// Proto_Metric_31937376813213
// MI455X (gfx1250) — compile-verified, matrix operand order checked by reading, not run on hardware
//
#include <hip/hip_runtime.h>

// ---------------- problem constants (from reference) ----------------
constexpr int Bq   = 8192;   // queries
constexpr int Ncls = 1000;   // classes
constexpr int NPAD = 1024;   // classes padded to multiple of 128
constexpr int SHOT = 5;
constexpr int Dk   = 2048;   // feature dim (K)

// ---------------- vector types ----------------
typedef _Float16 v16h __attribute__((ext_vector_type(16)));
typedef _Float16 v4h  __attribute__((ext_vector_type(4)));
typedef float    v8f  __attribute__((ext_vector_type(8)));
typedef float    v4f  __attribute__((ext_vector_type(4)));

// ---------------- kernel 1: prototypes (mean over shots) -> f16 hi/lo + p_sq ----
__global__ __launch_bounds__(256) void proto_prep_kernel(
    const float* __restrict__ x2, _Float16* __restrict__ phi,
    _Float16* __restrict__ plo, float* __restrict__ psq) {
  const int n = blockIdx.x;          // 0..NPAD-1
  const int t = threadIdx.x;
  float local = 0.0f;
  if (n < Ncls) {
    const float* base = x2 + (size_t)n * SHOT * Dk;
    for (int k = t; k < Dk; k += 256) {
      float s = 0.0f;
#pragma unroll
      for (int sh = 0; sh < SHOT; ++sh) s += base[sh * Dk + k];
      float m = s * 0.2f;
      _Float16 h = (_Float16)m;
      float r = m - (float)h;
      phi[(size_t)n * Dk + k] = h;
      plo[(size_t)n * Dk + k] = (_Float16)r;
      local += m * m;
    }
  } else {
    for (int k = t; k < Dk; k += 256) {
      phi[(size_t)n * Dk + k] = (_Float16)0.0f;
      plo[(size_t)n * Dk + k] = (_Float16)0.0f;
    }
  }
  __shared__ float red[256];
  red[t] = local;
  __syncthreads();
  for (int s = 128; s > 0; s >>= 1) {
    if (t < s) red[t] += red[t + s];
    __syncthreads();
  }
  if (t == 0) psq[n] = red[0];
}

// ---------------- kernel 2: q_sq per row (one wave32 per row) ----------------
__global__ __launch_bounds__(256) void qsq_kernel(const float* __restrict__ x1,
                                                  float* __restrict__ qsq) {
  const int t    = threadIdx.x;
  const int lane = t & 31;
  const int row  = blockIdx.x * 8 + (t >> 5);
  const v4f* p = (const v4f*)(x1 + (size_t)row * Dk);
  float s = 0.0f;
#pragma unroll
  for (int i = 0; i < Dk / 4 / 32; ++i) {
    v4f v = p[lane + i * 32];
    s += v[0] * v[0] + v[1] * v[1] + v[2] * v[2] + v[3] * v[3];
  }
#pragma unroll
  for (int off = 16; off > 0; off >>= 1) s += __shfl_xor(s, off, 32);
  if (lane == 0) qsq[row] = s;
}

// ---------------- kernel 3: WMMA GEMM + score epilogue ----------------
#define BM 128
#define BN 128
#define BK 32
#define LDSK 40              // 32 + 8 halves pad -> conflict-free b128 frags
#define TILEH (BM * LDSK)    // halves per sub-array (A-hi / A-lo / B-hi / B-lo)

__device__ __forceinline__ v8f wmma16(const v16h a, const v16h b, v8f c) {
  return __builtin_amdgcn_wmma_f32_16x16x32_f16(false, a, false, b, (short)0,
                                                c, false, false);
}

// async global -> LDS, 16 bytes per lane (GLOBAL_LOAD_ASYNC_TO_LDS_B128, GVS)
__device__ __forceinline__ void async_g2l_b128(const _Float16* gbase,
                                               unsigned goff_bytes,
                                               const _Float16* lds_dst) {
  unsigned lds_addr = (unsigned)(unsigned long long)lds_dst;  // addr[31:0]
  asm volatile("global_load_async_to_lds_b128 %0, %1, %2"
               :
               : "v"(lds_addr), "v"(goff_bytes), "s"(gbase)
               : "memory");
}

__device__ __forceinline__ void wait_async0() {
#if __has_builtin(__builtin_amdgcn_s_wait_asynccnt)
  __builtin_amdgcn_s_wait_asynccnt(0);
#else
  asm volatile("s_wait_asynccnt 0" ::: "memory");
#endif
}

__global__ __launch_bounds__(256) void proto_score_gemm_kernel(
    const float* __restrict__ x1, const _Float16* __restrict__ phi,
    const _Float16* __restrict__ plo, const float* __restrict__ psq,
    const float* __restrict__ qsq, float* __restrict__ out) {
  // double buffered: 2 x (A-hi, A-lo, B-hi, B-lo) = 80 KB
  __shared__ _Float16 smem[2 * 4 * TILEH];

  const int t    = threadIdx.x;
  const int lane = t & 31;
  const int w    = t >> 5;       // 0..7
  const int wm   = w & 1;        // 2 waves along M (64 rows each)
  const int wn   = w >> 1;       // 4 waves along N (32 cols each)
  const int bm   = blockIdx.x * BM;
  const int bn   = blockIdx.y * BN;

  const int rl  = lane & 15;            // row within 16x16 tile
  const int kof = (lane < 16) ? 0 : 8;  // K chunk offset per ISA A/B layout

  v8f acc[4][2] = {};
  v4f areg[4];                          // staged A (fp32) for next tile

  // ---- pipeline stage helpers ----
  auto issueA = [&](int k0) {
#pragma unroll
    for (int i = 0; i < 4; ++i) {
      int linear = t + i * 256;         // 0..1023 float4 slots of 128x32
      int row = linear >> 3;
      int kq  = (linear & 7) << 2;
      areg[i] = *(const v4f*)(x1 + (size_t)(bm + row) * Dk + k0 + kq);
    }
  };
  auto issueB = [&](int k0, int buf) {  // async f16 tiles straight into LDS
    _Float16* bHi = smem + buf * 4 * TILEH + 2 * TILEH;
    _Float16* bLo = bHi + TILEH;
#pragma unroll
    for (int i = 0; i < 2; ++i) {
      int linear = t + i * 256;         // 0..511 8-half slots of 128x32
      int row = linear >> 2;
      int kq  = (linear & 3) << 3;
      unsigned goff = (unsigned)(((bn + row) * Dk + k0 + kq) * 2);  // bytes
      async_g2l_b128(phi, goff, &bHi[row * LDSK + kq]);
      async_g2l_b128(plo, goff, &bLo[row * LDSK + kq]);
    }
  };
  auto stageA = [&](int buf) {          // fp32 -> hi/lo f16 into LDS
    _Float16* aHi = smem + buf * 4 * TILEH;
    _Float16* aLo = aHi + TILEH;
#pragma unroll
    for (int i = 0; i < 4; ++i) {
      int linear = t + i * 256;
      int row = linear >> 3;
      int kq  = (linear & 7) << 2;
      v4h h, l;
#pragma unroll
      for (int j = 0; j < 4; ++j) {
        _Float16 hh = (_Float16)areg[i][j];
        h[j] = hh;
        l[j] = (_Float16)(areg[i][j] - (float)hh);
      }
      *(v4h*)&aHi[row * LDSK + kq] = h;
      *(v4h*)&aLo[row * LDSK + kq] = l;
    }
  };
  auto compute = [&](int buf) {
    const _Float16* aHi = smem + buf * 4 * TILEH;
    const _Float16* aLo = aHi + TILEH;
    const _Float16* bHi = aLo + TILEH;
    const _Float16* bLo = bHi + TILEH;
    v16h ah[4], al[4], bh[2], bl[2];
#pragma unroll
    for (int mt = 0; mt < 4; ++mt) {
      int base = (wm * 64 + mt * 16 + rl) * LDSK + kof;
      ((v4f*)&ah[mt])[0] = *(const v4f*)&aHi[base];
      ((v4f*)&ah[mt])[1] = *(const v4f*)&aHi[base + 16];
      ((v4f*)&al[mt])[0] = *(const v4f*)&aLo[base];
      ((v4f*)&al[mt])[1] = *(const v4f*)&aLo[base + 16];
    }
#pragma unroll
    for (int nt = 0; nt < 2; ++nt) {
      int base = (wn * 32 + nt * 16 + rl) * LDSK + kof;
      ((v4f*)&bh[nt])[0] = *(const v4f*)&bHi[base];
      ((v4f*)&bh[nt])[1] = *(const v4f*)&bHi[base + 16];
      ((v4f*)&bl[nt])[0] = *(const v4f*)&bLo[base];
      ((v4f*)&bl[nt])[1] = *(const v4f*)&bLo[base + 16];
    }
#pragma unroll
    for (int mt = 0; mt < 4; ++mt) {
#pragma unroll
      for (int nt = 0; nt < 2; ++nt) {
        acc[mt][nt] = wmma16(ah[mt], bh[nt], acc[mt][nt]);
        acc[mt][nt] = wmma16(ah[mt], bl[nt], acc[mt][nt]);
        acc[mt][nt] = wmma16(al[mt], bh[nt], acc[mt][nt]);
      }
    }
  };

  // ---- prologue: fill buffer 0 ----
  issueA(0);
  issueB(0, 0);
  stageA(0);
  wait_async0();
  __syncthreads();

  // ---- main pipelined loop: one barrier per K-step ----
  constexpr int KITER = Dk / BK;  // 64
  int buf = 0;
  for (int k = 0; k < KITER; ++k) {
    const int nxt = buf ^ 1;
    if (k + 1 < KITER) {
      issueB((k + 1) * BK, nxt);   // async B for next tile (overlaps WMMA)
      issueA((k + 1) * BK);        // global A loads in flight during WMMA
    }
    compute(buf);
    if (k + 1 < KITER) {
      stageA(nxt);                 // convert + ds_store after compute
      wait_async0();               // B tile landed in LDS
    }
    __syncthreads();               // publish buf^1, retire reads of buf
    buf = nxt;
  }

  // ---- epilogue: score = 2*cross - q_sq - p_sq ----
  int   col[2];
  float ps[2];
#pragma unroll
  for (int nt = 0; nt < 2; ++nt) {
    col[nt] = bn + wn * 32 + nt * 16 + rl;
    ps[nt]  = psq[col[nt]];   // padded to NPAD, always in-bounds
  }
  const int mhi = (lane >= 16) ? 8 : 0;
#pragma unroll
  for (int mt = 0; mt < 4; ++mt) {
#pragma unroll
    for (int v = 0; v < 8; ++v) {
      int row = bm + wm * 64 + mt * 16 + v + mhi;
      float qs = qsq[row];
#pragma unroll
      for (int nt = 0; nt < 2; ++nt) {
        if (col[nt] < Ncls)
          out[(size_t)row * Ncls + col[nt]] =
              2.0f * acc[mt][nt][v] - qs - ps[nt];
      }
    }
  }
}

// ---------------- launcher ----------------
extern "C" void kernel_launch(void* const* d_in, const int* in_sizes, int n_in,
                              void* d_out, int out_size, void* d_ws,
                              size_t ws_size, hipStream_t stream) {
  (void)in_sizes; (void)n_in; (void)out_size; (void)ws_size;
  const float* x1 = (const float*)d_in[0];   // [B, D]
  const float* x2 = (const float*)d_in[1];   // [N, SHOT, D]
  float* out = (float*)d_out;                // [B, N]

  // workspace layout (~8.4 MB)
  _Float16* phi = (_Float16*)d_ws;                       // [NPAD, D] f16
  _Float16* plo = phi + (size_t)NPAD * Dk;               // [NPAD, D] f16
  float*    psq = (float*)(plo + (size_t)NPAD * Dk);     // [NPAD]
  float*    qsq = psq + NPAD;                            // [B]

  proto_prep_kernel<<<NPAD, 256, 0, stream>>>(x2, phi, plo, psq);
  qsq_kernel<<<Bq / 8, 256, 0, stream>>>(x1, qsq);
  dim3 grid(Bq / BM, NPAD / BN);
  proto_score_gemm_kernel<<<grid, 256, 0, stream>>>(x1, phi, plo, psq, qsq, out);
}
